// CNN_Concat_6828998001213
// MI455X (gfx1250) — compile-verified, matrix operand order checked by reading, not run on hardware
//
#include <hip/hip_runtime.h>
#include <hip/hip_bf16.h>

// ============================================================================
// CNN_Concat model for MI455X (gfx1250, wave32, WMMA).
//
//   embed -> [GEMM gXf = X@Wih_f^T + b]  -> persistent-LSTM(front)
//   conv blocks (4x residual 3D conv + batchnorm)  -> depth mean
//   concat(z, lstm_front) -> [GEMM gate] -> sigmoid mix
//   [GEMM gXr = mixed@Wih_r^T + b] -> persistent-LSTM(rear)
//   [GEMM out = rear@Wf^T + bf] -> d_out
//
// All dense matmuls use v_wmma_f32_16x16x32_bf16 (bf16 in, f32 accum).
// All K and N dims are zero-padded to full tile groups so the WMMA loops are
// completely branch-free (EXEC all-ones, no phi/acc-copy churn).
// Input index map assumes jax pytree (sorted dict key) flattening -- see enum.
// ============================================================================

#define B_   16
#define T_   200
#define H_   15
#define D_   225            // H*H == HID
#define HID_ 225
#define OUT_ 110
#define BT_  (B_ * T_)      // 3200
#define G4_  900            // 4*HID
#define GLD_ 1024           // gate row stride, padded to 64 N-tiles
#define NTG_ 64             // 64 N-tiles (= GLD_/16) for LSTM gate GEMMs
#define KP_  256            // K pad for 225 (8 chunks of 32)
#define KC_D 8
#define KP_GATE 480         // K pad for 450 (15 chunks of 32)
#define KC_GATE 15
#define VOX_ 1800           // 8*15*15 voxels per clip per channel
#define EPS_ 1e-5f

typedef __attribute__((ext_vector_type(16))) __bf16 v16bf;
typedef __attribute__((ext_vector_type(8)))  __bf16 v8bf;
typedef __attribute__((ext_vector_type(8)))  float  v8f;

__device__ __forceinline__ __bf16 f2bf(float f) {
  union { float f; unsigned u; } v; v.f = f;
  unsigned r = (v.u + 0x7FFFu + ((v.u >> 16) & 1u)) >> 16;
  unsigned short s = (unsigned short)r;
  __bf16 out; __builtin_memcpy(&out, &s, 2); return out;
}

__device__ __forceinline__ float sigf(float x) { return 1.0f / (1.0f + __expf(-x)); }

// Load one 16x32 bf16 fragment (A layout; B uses the same layout on the
// row-major (N,K) weight matrix). Per CDNA5 ISA 7.12.2:
//   lanes 0-15 : row = lane,    K = {0..7, 16..23}
//   lanes 16-31: row = lane-16, K = {8..15, 24..31}
// ld must be a multiple of 8 bf16 elements (16B alignment).
__device__ __forceinline__ v16bf load_frag(const __bf16* base, int ld, int lane) {
  int m  = lane & 15;
  int kh = (lane >> 4) << 3;               // 0 or 8
  const __bf16* p = base + (size_t)m * ld + kh;
  v8bf lo = *(const v8bf*)(p);
  v8bf hi = *(const v8bf*)(p + 16);
  v16bf r;
#pragma unroll
  for (int i = 0; i < 8; ++i) { r[i] = lo[i]; r[i + 8] = hi[i]; }
  return r;
}

// ---------------------------------------------------------------------------
// Tiled WMMA GEMM: C[M,N] = A(M x K bf16 row-major, lda) x B^T + bias[n],
// B given as (N x K bf16 row-major, ldb), N padded to gridDim.x*64 rows.
// Block = 128 threads (4 waves); wave w -> M-tile blockIdx.y*4+w; each wave
// computes 4 consecutive N-tiles unconditionally (A fragment reused 4x).
// KC (k-chunks of 32) is a template constant so the loop fully unrolls.
// ---------------------------------------------------------------------------
template <int KC>
__global__ __launch_bounds__(128) void gemm_bf16_wmma(
    const __bf16* __restrict__ A, int lda,
    const __bf16* __restrict__ Bm, int ldb,
    float* __restrict__ C, int ldc,
    const float* __restrict__ bias,
    int nStoreMax)
{
  int lane = threadIdx.x & 31;
  int wave = threadIdx.x >> 5;
  int mTile = blockIdx.y * 4 + wave;
  int nt0 = blockIdx.x * 4;
  const __bf16* Arow = A + (size_t)mTile * 16 * lda;
  const __bf16* Brow = Bm + (size_t)nt0 * 16 * ldb;

  v8f acc0 = {}, acc1 = {}, acc2 = {}, acc3 = {};
#pragma unroll
  for (int kc = 0; kc < KC; ++kc) {
    __builtin_prefetch(Arow + (kc + 1) * 32, 0, 1);          // global_prefetch_b8
    v16bf a  = load_frag(Arow + kc * 32, lda, lane);
    v16bf b0 = load_frag(Brow + (size_t)0 * 16 * ldb + kc * 32, ldb, lane);
    v16bf b1 = load_frag(Brow + (size_t)1 * 16 * ldb + kc * 32, ldb, lane);
    v16bf b2 = load_frag(Brow + (size_t)2 * 16 * ldb + kc * 32, ldb, lane);
    v16bf b3 = load_frag(Brow + (size_t)3 * 16 * ldb + kc * 32, ldb, lane);
    acc0 = __builtin_amdgcn_wmma_f32_16x16x32_bf16(false, a, false, b0, (short)0, acc0, false, false);
    acc1 = __builtin_amdgcn_wmma_f32_16x16x32_bf16(false, a, false, b1, (short)0, acc1, false, false);
    acc2 = __builtin_amdgcn_wmma_f32_16x16x32_bf16(false, a, false, b2, (short)0, acc2, false, false);
    acc3 = __builtin_amdgcn_wmma_f32_16x16x32_bf16(false, a, false, b3, (short)0, acc3, false, false);
  }

  // C/D layout: VGPR i, lanes 0-15 -> (M=i, N=lane); lanes 16-31 -> (M=i+8).
  int n = lane & 15, half = lane >> 4;
  v8f accs[4] = {acc0, acc1, acc2, acc3};
#pragma unroll
  for (int j = 0; j < 4; ++j) {
    int col = (nt0 + j) * 16 + n;
    if (col >= nStoreMax) continue;                          // epilogue-only guard
    float bv = bias ? bias[col] : 0.0f;
#pragma unroll
    for (int i = 0; i < 8; ++i) {
      int row = mTile * 16 + half * 8 + i;
      C[(size_t)row * ldc + col] = accs[j][i] + bv;
    }
  }
}

// ---------------------------------------------------------------------------
// Persistent sequential LSTM. One workgroup of 1024 threads (32 waves).
// gX = x@Wih^T + (bih+bhh) precomputed, rows (b*T + t), stride GLD_=1024.
// Per step: gates = gX[t] + h@Whh^T via 64 WMMA N-tiles (2 per wave, fully
// unrolled), then the elementwise cell. h kept bf16-padded (16x256) in LDS.
// ---------------------------------------------------------------------------
template <bool HSOUT, bool HBF>
__global__ __launch_bounds__(1024, 1) void lstm_seq(
    const float* __restrict__ gX,       // (B*T, 1024)
    const __bf16* __restrict__ Whh,     // (1024, 256) zero-padded bf16
    const float* __restrict__ h0,       // (16, 225)
    const float* __restrict__ c0,       // (16, 225)
    float* __restrict__ hsOut,          // (B*T, 225)   if HSOUT
    __bf16* __restrict__ hbfOut)        // (B*T, 256)   if HBF
{
  __shared__ __align__(16) __bf16 hsh[16 * KP_];
  __shared__ float csh[16 * HID_];
  __shared__ float gsh[16 * GLD_];

  int tid = threadIdx.x;
  for (int idx = tid; idx < 16 * KP_; idx += blockDim.x) {
    int m = idx >> 8, j = idx & 255;
    hsh[idx] = f2bf(j < HID_ ? h0[m * HID_ + j] : 0.0f);
  }
  for (int idx = tid; idx < 16 * HID_; idx += blockDim.x) csh[idx] = c0[idx];
  __syncthreads();

  int lane = tid & 31, wave = tid >> 5;

  for (int t = 0; t < T_; ++t) {
    // ---- recurrence GEMM: gates = gX(t) + h @ Whh^T (64 tiles, 2/wave) ----
#pragma unroll
    for (int rep = 0; rep < 2; ++rep) {
      int nt = wave + rep * 32;
      v8f acc = {};
#pragma unroll
      for (int kc = 0; kc < KC_D; ++kc) {
        v16bf a = load_frag(hsh + kc * 32, KP_, lane);
        v16bf b = load_frag(Whh + (size_t)nt * 16 * KP_ + kc * 32, KP_, lane);
        acc = __builtin_amdgcn_wmma_f32_16x16x32_bf16(
            false, a, false, b, (short)0, acc, false, false);
      }
      int n = lane & 15, half = lane >> 4;
      int col = nt * 16 + n;
#pragma unroll
      for (int i = 0; i < 8; ++i) {
        int m = half * 8 + i;
        gsh[m * GLD_ + col] = acc[i] + gX[((size_t)m * T_ + t) * GLD_ + col];
      }
    }
    __syncthreads();
    // ---- elementwise cell -------------------------------------------------
    for (int idx = tid; idx < 16 * HID_; idx += blockDim.x) {
      int m = idx / HID_, j = idx % HID_;
      const float* g = gsh + m * GLD_;
      float i_ = g[j];
      float f_ = g[HID_ + j];
      float gg = g[2 * HID_ + j];
      float o_ = g[3 * HID_ + j];
      float c = sigf(f_) * csh[idx] + sigf(i_) * tanhf(gg);
      float h = sigf(o_) * tanhf(c);
      csh[idx] = c;
      hsh[m * KP_ + j] = f2bf(h);
      size_t row = (size_t)m * T_ + t;
      if (HSOUT) hsOut[row * HID_ + j] = h;
      if (HBF)   hbfOut[row * KP_ + j] = f2bf(h);
    }
    if (HBF) {                          // zero pad cols 225..255
      for (int idx = tid; idx < 16 * (KP_ - HID_); idx += blockDim.x) {
        int m = idx / (KP_ - HID_), j = HID_ + idx % (KP_ - HID_);
        hbfOut[((size_t)m * T_ + t) * KP_ + j] = f2bf(0.0f);
      }
    }
    __syncthreads();
  }
}

// ---------------------------------------------------------------------------
// Utility kernels
// ---------------------------------------------------------------------------
__global__ void zero_f32(float* p, int n) {
  int i = blockIdx.x * blockDim.x + threadIdx.x;
  if (i < n) p[i] = 0.0f;
}

// dst (drows x dcols bf16) = zero-padded copy of src (rows x cols f32)
__global__ void pad_convert(const float* __restrict__ src, int rows, int cols,
                            __bf16* __restrict__ dst, int drows, int dcols) {
  int idx = blockIdx.x * blockDim.x + threadIdx.x;
  if (idx >= drows * dcols) return;
  int r = idx / dcols, c = idx % dcols;
  float v = (r < rows && c < cols) ? src[(size_t)r * cols + c] : 0.0f;
  dst[idx] = f2bf(v);
}

__global__ void bias_pad(const float* __restrict__ a, const float* __restrict__ b,
                         float* __restrict__ out, int n, int npad) {
  int i = blockIdx.x * blockDim.x + threadIdx.x;
  if (i >= npad) return;
  out[i] = (i < n) ? (a[i] + (b ? b[i] : 0.0f)) : 0.0f;
}

// Embedding gather: embF (3200x225 f32) and padded Xbf (3200x256 bf16).
__global__ void embed_kernel(const int* __restrict__ x, const float* __restrict__ emb,
                             float* __restrict__ embF, __bf16* __restrict__ Xbf) {
  int idx = blockIdx.x * blockDim.x + threadIdx.x;
  if (idx >= BT_ * KP_) return;
  int r = idx >> 8, j = idx & 255;
  int tok = x[r];
  float v = (j < D_) ? emb[(size_t)tok * D_ + j] : 0.0f;
  if (j < D_) embF[(size_t)r * D_ + j] = v;
  Xbf[idx] = f2bf(v);
}

// ---------------------------------------------------------------------------
// Direct 3D conv (ks=3 pad=1, or ks=1 pad=0), output (N, CO, 8, 15, 15),
// with fused per-channel sum/sumsq reduction for batchnorm.
// X == nullptr -> "clips" mode: channel 0 gathered from emb frames with
// causal time padding (clip depth d maps to frame t-7+d).
// ---------------------------------------------------------------------------
__global__ __launch_bounds__(256) void conv3d_stats(
    const float* __restrict__ X, const float* __restrict__ embF,
    const float* __restrict__ W,
    float* __restrict__ Y, float* __restrict__ stats,
    int CI, int CO, int ks, int pad)
{
  int co = blockIdx.y;
  int idx = blockIdx.x * blockDim.x + threadIdx.x;     // over N*1800
  float val = 0.0f;
  if (idx < BT_ * VOX_) {
    int n = idx / VOX_;
    int rem = idx % VOX_;
    int d = rem / 225, hw = rem % 225, yy = hw / 15, xx = hw % 15;
    float acc = 0.0f;
    for (int ci = 0; ci < CI; ++ci) {
      for (int kd = 0; kd < ks; ++kd) {
        int id = d + kd - pad; if ((unsigned)id >= 8u) continue;
        for (int kh = 0; kh < ks; ++kh) {
          int iy = yy + kh - pad; if ((unsigned)iy >= 15u) continue;
          for (int kw = 0; kw < ks; ++kw) {
            int ix = xx + kw - pad; if ((unsigned)ix >= 15u) continue;
            float xv;
            if (X) {
              xv = X[(((size_t)n * CI + ci) * 8 + id) * 225 + iy * 15 + ix];
            } else {
              int b = n / T_, t = n % T_;
              int ft = t - 7 + id;                      // K-1 = 7 left pad
              xv = (ft >= 0) ? embF[((size_t)b * T_ + ft) * D_ + iy * 15 + ix]
                             : 0.0f;
            }
            acc += xv * W[((((size_t)co * CI + ci) * ks + kd) * ks + kh) * ks + kw];
          }
        }
      }
    }
    Y[(((size_t)n * CO + co) * VOX_) + rem] = acc;
    val = acc;
  }
  // block reduction of (sum, sumsq) -> 2 global atomics per block
  __shared__ float s1[256], s2[256];
  int tid = threadIdx.x;
  s1[tid] = val; s2[tid] = val * val;
  __syncthreads();
  for (int off = 128; off > 0; off >>= 1) {
    if (tid < off) { s1[tid] += s1[tid + off]; s2[tid] += s2[tid + off]; }
    __syncthreads();
  }
  if (tid == 0) {
    atomicAdd(&stats[co * 2 + 0], s1[0]);
    atomicAdd(&stats[co * 2 + 1], s2[0]);
  }
}

// In-place batchnorm (+ optional relu) on (N, CO, 1800) buffer.
__global__ void bn_act(float* __restrict__ Y, const float* __restrict__ stats,
                       const float* __restrict__ g, const float* __restrict__ b,
                       int CO, int relu) {
  int total = BT_ * CO * VOX_;
  int idx = blockIdx.x * blockDim.x + threadIdx.x;
  if (idx >= total) return;
  int co = (idx / VOX_) % CO;
  const float cnt = (float)BT_ * (float)VOX_;
  float mean = stats[co * 2] / cnt;
  float var  = stats[co * 2 + 1] / cnt - mean * mean;
  float v = (Y[idx] - mean) * rsqrtf(var + EPS_) * g[co] + b[co];
  Y[idx] = relu ? fmaxf(v, 0.0f) : v;
}

// Y = relu( bn(T2) + bn(T3) )  -- residual merge of a block.
__global__ void bn_add_relu(const float* __restrict__ T2, const float* __restrict__ s2,
                            const float* __restrict__ g2, const float* __restrict__ b2,
                            const float* __restrict__ T3, const float* __restrict__ sd,
                            const float* __restrict__ gd, const float* __restrict__ bd,
                            float* __restrict__ Y, int CO) {
  int total = BT_ * CO * VOX_;
  int idx = blockIdx.x * blockDim.x + threadIdx.x;
  if (idx >= total) return;
  int co = (idx / VOX_) % CO;
  const float cnt = (float)BT_ * (float)VOX_;
  float m2 = s2[co * 2] / cnt, v2 = s2[co * 2 + 1] / cnt - m2 * m2;
  float md = sd[co * 2] / cnt, vd = sd[co * 2 + 1] / cnt - md * md;
  float a = (T2[idx] - m2) * rsqrtf(v2 + EPS_) * g2[co] + b2[co];
  float r = (T3[idx] - md) * rsqrtf(vd + EPS_) * gd[co] + bd[co];
  Y[idx] = fmaxf(a + r, 0.0f);
}

// z.mean(axis=depth): (3200, 1, 8, 15, 15) -> (3200, 225)
__global__ void depth_mean(const float* __restrict__ Z, float* __restrict__ out) {
  int idx = blockIdx.x * blockDim.x + threadIdx.x;
  if (idx >= BT_ * D_) return;
  int n = idx / D_, hw = idx % D_;
  float s = 0.0f;
#pragma unroll
  for (int d = 0; d < 8; ++d) s += Z[(size_t)n * VOX_ + d * 225 + hw];
  out[idx] = s * 0.125f;
}

// concat(z, lstm_front) -> bf16 (3200 x 480), K-padded with zeros.
__global__ void build_concat(const float* __restrict__ z, const float* __restrict__ hF,
                             __bf16* __restrict__ G) {
  int idx = blockIdx.x * blockDim.x + threadIdx.x;
  if (idx >= BT_ * KP_GATE) return;
  int r = idx / KP_GATE, j = idx % KP_GATE;
  float v = (j < D_) ? z[(size_t)r * D_ + j]
          : (j < 2 * D_) ? hF[(size_t)r * D_ + (j - D_)] : 0.0f;
  G[idx] = f2bf(v);
}

// u = sigmoid(gateLin); mixed = u*z + (1-u)*lstm_out  -> bf16 padded (x256)
__global__ void gate_mix(const float* __restrict__ gateLin, const float* __restrict__ z,
                         const float* __restrict__ hF, __bf16* __restrict__ mixedBf) {
  int idx = blockIdx.x * blockDim.x + threadIdx.x;
  if (idx >= BT_ * KP_) return;
  int r = idx >> 8, j = idx & 255;
  float v = 0.0f;
  if (j < HID_) {
    float u = sigf(gateLin[(size_t)r * 240 + j]);
    v = u * z[(size_t)r * D_ + j] + (1.0f - u) * hF[(size_t)r * D_ + j];
  }
  mixedBf[idx] = f2bf(v);
}

// ============================================================================
// Host side
// ============================================================================
static inline char* wsAlloc(char*& p, size_t bytes) {
  char* r = p;
  p += (bytes + 255) & ~(size_t)255;
  return r;
}

// jax-pytree (sorted key) flattening of setup_inputs() = {'params':..., 'x':...}
enum {
  IN_Wf = 0, IN_Wg = 1, IN_bf = 2, IN_bg = 3,
  IN_BLK0 = 4,            // per block (9): b1,b2,bd,g1,g2,gd,w1,w2,wd
  IN_EMB = 40,
  IN_F_Whh = 41, IN_F_Wih = 42, IN_F_bhh = 43, IN_F_bih = 44, IN_F_c0 = 45, IN_F_h0 = 46,
  IN_R_Whh = 47, IN_R_Wih = 48, IN_R_bhh = 49, IN_R_bih = 50, IN_R_c0 = 51, IN_R_h0 = 52,
  IN_X = 53
};

extern "C" void kernel_launch(void* const* d_in, const int* in_sizes, int n_in,
                              void* d_out, int out_size, void* d_ws, size_t ws_size,
                              hipStream_t stream) {
  (void)in_sizes; (void)n_in; (void)out_size; (void)ws_size;

  // ---- workspace layout (~790 MB) -----------------------------------------
  char* p = (char*)d_ws;
  float*  stats   = (float*)wsAlloc(p, 12 * 32 * sizeof(float));
  __bf16* WihF    = (__bf16*)wsAlloc(p, (size_t)GLD_ * KP_ * 2);      // 1024x256
  __bf16* WhhF    = (__bf16*)wsAlloc(p, (size_t)GLD_ * KP_ * 2);
  __bf16* WihR    = (__bf16*)wsAlloc(p, (size_t)GLD_ * KP_ * 2);
  __bf16* WhhR    = (__bf16*)wsAlloc(p, (size_t)GLD_ * KP_ * 2);
  __bf16* WgB     = (__bf16*)wsAlloc(p, (size_t)256 * KP_GATE * 2);   // 16 tiles
  __bf16* WfB     = (__bf16*)wsAlloc(p, (size_t)128 * KP_ * 2);       // 8 tiles
  float*  ballF   = (float*)wsAlloc(p, GLD_ * sizeof(float));
  float*  ballR   = (float*)wsAlloc(p, GLD_ * sizeof(float));
  float*  bgP     = (float*)wsAlloc(p, 256 * sizeof(float));
  float*  bfP     = (float*)wsAlloc(p, 128 * sizeof(float));
  float*  embF    = (float*)wsAlloc(p, (size_t)BT_ * D_ * sizeof(float));
  __bf16* Xbf     = (__bf16*)wsAlloc(p, (size_t)BT_ * KP_ * 2);
  float*  gXf     = (float*)wsAlloc(p, (size_t)BT_ * GLD_ * sizeof(float));
  float*  hsF     = (float*)wsAlloc(p, (size_t)BT_ * D_ * sizeof(float));
  size_t  poolSz  = (size_t)BT_ * 8 * VOX_ * sizeof(float);
  float*  P0      = (float*)wsAlloc(p, poolSz);
  float*  P1      = (float*)wsAlloc(p, poolSz);
  float*  P2      = (float*)wsAlloc(p, poolSz);
  float*  P3      = (float*)wsAlloc(p, poolSz);
  float*  zmean   = (float*)wsAlloc(p, (size_t)BT_ * D_ * sizeof(float));
  __bf16* concatG = (__bf16*)wsAlloc(p, (size_t)BT_ * KP_GATE * 2);
  float*  gateLin = (float*)wsAlloc(p, (size_t)BT_ * 240 * sizeof(float));
  __bf16* mixedBf = (__bf16*)wsAlloc(p, (size_t)BT_ * KP_ * 2);
  float*  gXr     = (float*)wsAlloc(p, (size_t)BT_ * GLD_ * sizeof(float));
  __bf16* hsRbf   = (__bf16*)wsAlloc(p, (size_t)BT_ * KP_ * 2);

  auto F = [&](int i) { return (const float*)d_in[i]; };

  const int TPB = 256;
  auto grids = [&](long n) { return dim3((unsigned)((n + TPB - 1) / TPB)); };

  // ---- stage 0: stats zero + weight conversion ----------------------------
  zero_f32<<<grids(12 * 32), TPB, 0, stream>>>(stats, 12 * 32);
  pad_convert<<<grids((long)GLD_ * KP_), TPB, 0, stream>>>(F(IN_F_Wih), G4_, D_, WihF, GLD_, KP_);
  pad_convert<<<grids((long)GLD_ * KP_), TPB, 0, stream>>>(F(IN_F_Whh), G4_, HID_, WhhF, GLD_, KP_);
  pad_convert<<<grids((long)GLD_ * KP_), TPB, 0, stream>>>(F(IN_R_Wih), G4_, HID_, WihR, GLD_, KP_);
  pad_convert<<<grids((long)GLD_ * KP_), TPB, 0, stream>>>(F(IN_R_Whh), G4_, HID_, WhhR, GLD_, KP_);
  pad_convert<<<grids((long)256 * KP_GATE), TPB, 0, stream>>>(F(IN_Wg), HID_, 2 * HID_, WgB, 256, KP_GATE);
  pad_convert<<<grids((long)128 * KP_), TPB, 0, stream>>>(F(IN_Wf), OUT_, HID_, WfB, 128, KP_);
  bias_pad<<<grids(GLD_), TPB, 0, stream>>>(F(IN_F_bih), F(IN_F_bhh), ballF, G4_, GLD_);
  bias_pad<<<grids(GLD_), TPB, 0, stream>>>(F(IN_R_bih), F(IN_R_bhh), ballR, G4_, GLD_);
  bias_pad<<<grids(256), TPB, 0, stream>>>(F(IN_bg), nullptr, bgP, HID_, 256);
  bias_pad<<<grids(128), TPB, 0, stream>>>(F(IN_bf), nullptr, bfP, OUT_, 128);

  // ---- stage 1: embedding --------------------------------------------------
  embed_kernel<<<grids((long)BT_ * KP_), TPB, 0, stream>>>(
      (const int*)d_in[IN_X], F(IN_EMB), embF, Xbf);

  const unsigned MGRP = BT_ / 16 / 4;   // 50 blocks of 4 M-tiles
  // ---- stage 2: front LSTM -------------------------------------------------
  gemm_bf16_wmma<KC_D><<<dim3(NTG_ / 4, MGRP), 128, 0, stream>>>(
      Xbf, KP_, WihF, KP_, gXf, GLD_, ballF, GLD_);
  lstm_seq<true, false><<<1, 1024, 0, stream>>>(
      gXf, WhhF, F(IN_F_h0), F(IN_F_c0), hsF, nullptr);

  // ---- stage 3: conv blocks ------------------------------------------------
  auto BP = [&](int blk, int k) { return (const float*)d_in[IN_BLK0 + 9 * blk + k]; };
  const int CIs[4] = {1, 4, 8, 4}, COs[4] = {4, 8, 4, 1};
  // pool assignment (X stays live until convd):  blk: X    t1  t2  t3  Y
  float* T1s[4] = {P0, P0, P1, P2};               // 0: emb  P0  P1  P2  P3
  float* T2s[4] = {P1, P1, P3, P3};               // 1: P3   P0  P1  P2  P0
  float* T3s[4] = {P2, P2, P2, P0};               // 2: P0   P1  P3  P2  P1
  float* Ys [4] = {P3, P0, P1, P2};               // 3: P1   P2  P3  P0  P2

  const float* Xcur = nullptr;                    // clips mode for block 0
  for (int blk = 0; blk < 4; ++blk) {
    int CI = CIs[blk], CO = COs[blk];
    float* s1 = stats + (blk * 3 + 0) * 32;
    float* s2 = stats + (blk * 3 + 1) * 32;
    float* sd = stats + (blk * 3 + 2) * 32;
    dim3 cg((unsigned)((BT_ * VOX_ + TPB - 1) / TPB), (unsigned)CO);
    conv3d_stats<<<cg, TPB, 0, stream>>>(Xcur, embF, BP(blk, 6), T1s[blk], s1, CI, CO, 3, 1);
    bn_act<<<grids((long)BT_ * CO * VOX_), TPB, 0, stream>>>(T1s[blk], s1, BP(blk, 3), BP(blk, 0), CO, 1);
    conv3d_stats<<<cg, TPB, 0, stream>>>(T1s[blk], embF, BP(blk, 7), T2s[blk], s2, CO, CO, 3, 1);
    conv3d_stats<<<cg, TPB, 0, stream>>>(Xcur, embF, BP(blk, 8), T3s[blk], sd, CI, CO, 1, 0);
    bn_add_relu<<<grids((long)BT_ * CO * VOX_), TPB, 0, stream>>>(
        T2s[blk], s2, BP(blk, 4), BP(blk, 1),
        T3s[blk], sd, BP(blk, 5), BP(blk, 2), Ys[blk], CO);
    Xcur = Ys[blk];
  }
  const float* zfinal = Ys[3];          // (3200, 1, 8, 15, 15)

  // ---- stage 4: depth mean + gate mix -------------------------------------
  depth_mean<<<grids((long)BT_ * D_), TPB, 0, stream>>>(zfinal, zmean);
  build_concat<<<grids((long)BT_ * KP_GATE), TPB, 0, stream>>>(zmean, hsF, concatG);
  gemm_bf16_wmma<KC_GATE><<<dim3(4, MGRP), 128, 0, stream>>>(
      concatG, KP_GATE, WgB, KP_GATE, gateLin, 240, bgP, 240);
  gate_mix<<<grids((long)BT_ * KP_), TPB, 0, stream>>>(gateLin, zmean, hsF, mixedBf);

  // ---- stage 5: rear LSTM --------------------------------------------------
  gemm_bf16_wmma<KC_D><<<dim3(NTG_ / 4, MGRP), 128, 0, stream>>>(
      mixedBf, KP_, WihR, KP_, gXr, GLD_, ballR, GLD_);
  lstm_seq<false, true><<<1, 1024, 0, stream>>>(
      gXr, WhhR, F(IN_R_h0), F(IN_R_c0), nullptr, hsRbf);

  // ---- stage 6: output projection -----------------------------------------
  gemm_bf16_wmma<KC_D><<<dim3(2, MGRP), 128, 0, stream>>>(
      hsRbf, KP_, WfB, KP_, (float*)d_out, OUT_, bfP, OUT_);
}
